// Decoder_8486855377266
// MI455X (gfx1250) — matrix operand order repaired and checked by reading, not compiled, not run on hardware
//
#include <hip/hip_runtime.h>
#include <hip/hip_bf16.h>

#define BB 2
#define TT 2048
#define VV 32000
#define CC 1024
#define HH 16
#define HSZ 64
#define BT (BB * TT)

typedef __bf16 bf16;
typedef __attribute__((ext_vector_type(16))) __bf16 v16bf;
typedef __attribute__((ext_vector_type(8)))  __bf16 v8bf;
typedef __attribute__((ext_vector_type(8)))  float  v8f;

// ---------------------------------------------------------------------------
// WMMA helper: D = A(16x32 bf16) * B(32x16 bf16) + C(16x16 f32)
// ---------------------------------------------------------------------------
__device__ __forceinline__ v8f wmma_acc(v16bf a, v16bf b, v8f c) {
    return __builtin_amdgcn_wmma_f32_16x16x32_bf16(
        /*neg_a=*/false, a, /*neg_b=*/false, b,
        /*c_mod=*/(short)0, c, /*reuse_a=*/false, /*reuse_b=*/false);
}

// A-fragment (16x32, row-major source, leading dim lda elements).
// Layout: lanes 0-15 -> M=lane, K in {0..15}; lanes 16-31 -> M=lane-16,
// K in {16..31}. Each half is 8 contiguous bf16 (16B aligned).
__device__ __forceinline__ v16bf load_a16x32(const bf16* base, int lda) {
    int lane = threadIdx.x & 31;
    const bf16* p = base + (long)(lane & 15) * lda + ((lane >> 4) << 4);
    v8bf lo = *(const v8bf*)p;
    v8bf hi = *(const v8bf*)(p + 8);
    v16bf a;
#pragma unroll
    for (int i = 0; i < 8; ++i) { a[i] = lo[i]; a[8 + i] = hi[i]; }
    return a;
}

// B-fragment (32x16) from a TRANSPOSED source: baseT is [N][K] row-major with
// leading dim ldk, so each lane reads 16 contiguous K values (one 32B load).
// Contiguous-K order used here (same as A): lane = N (mod 16), lanes 0-15 hold K=0..15, lanes 16-31 K=16..31.
__device__ __forceinline__ v16bf load_b32x16_t(const bf16* baseT, int ldk) {
    int lane = threadIdx.x & 31;
    const bf16* p = baseT + (long)(lane & 15) * ldk + ((lane >> 4) << 4);
    return *(const v16bf*)p;
}

// ---------------------------------------------------------------------------
// 1) Embedding: h[b,t,:] = tok_emb[x[b,t]] + pos_emb[t]  (-> bf16)
// ---------------------------------------------------------------------------
__global__ void k_embed(const int* __restrict__ x, const float* __restrict__ tok,
                        const float* __restrict__ pos, bf16* __restrict__ h) {
    int row = blockIdx.x;              // 0..BT-1
    int t = row & (TT - 1);
    long tk = (long)x[row];
    const float* te = tok + tk * CC;
    const float* pe = pos + (long)t * CC;
    bf16* hr = h + (long)row * CC;
    for (int c = threadIdx.x; c < CC; c += blockDim.x)
        hr[c] = (bf16)(te[c] + pe[c]);
}

// ---------------------------------------------------------------------------
// 2) Cast+transpose QKV weight:  [H][C][HS] f32 -> [H][HS][C] bf16
// ---------------------------------------------------------------------------
__global__ void k_cast_wqkv(const float* __restrict__ w, bf16* __restrict__ wt) {
    long i = (long)blockIdx.x * blockDim.x + threadIdx.x;   // over H*HS*C
    if (i >= (long)HH * HSZ * CC) return;
    int c = (int)(i % CC);
    long rem = i / CC;
    int d = (int)(rem % HSZ);
    int hh = (int)(rem / HSZ);
    wt[i] = (bf16)(w[((long)hh * CC + c) * HSZ + d]);
}

// ---------------------------------------------------------------------------
// 3) Cast+transpose head weight:  [C][V] f32 -> [V][C] bf16  (LDS tiled)
// ---------------------------------------------------------------------------
__global__ void k_cast_whead(const float* __restrict__ w, bf16* __restrict__ wt) {
    __shared__ float tile[64][65];
    int tv = blockIdx.x;   // V tile
    int tc = blockIdx.y;   // C tile
    for (int l = threadIdx.x; l < 64 * 64; l += blockDim.x) {
        int r = l >> 6, c = l & 63;     // r: C-dir, c: V-dir
        tile[r][c] = w[(long)(tc * 64 + r) * VV + tv * 64 + c];
    }
    __syncthreads();
    for (int l = threadIdx.x; l < 64 * 64; l += blockDim.x) {
        int r = l >> 6, c = l & 63;     // r: V-dir, c: C-dir
        wt[(long)(tv * 64 + r) * CC + tc * 64 + c] = (bf16)tile[c][r];
    }
}

// ---------------------------------------------------------------------------
// 4) QKV projection via WMMA. One wave per 16x16 output tile.
//    q,k -> [B][H][T][HS] bf16 ;  v -> transposed [B][H][HS][T] bf16.
// ---------------------------------------------------------------------------
__global__ void k_qkv(const bf16* __restrict__ h, const bf16* __restrict__ wt,
                      const float* __restrict__ bq, const float* __restrict__ bk,
                      const float* __restrict__ bv,
                      bf16* __restrict__ q, bf16* __restrict__ k,
                      bf16* __restrict__ vT) {
    int mt = blockIdx.x;               // 16-row tile of BT
    int nt = blockIdx.y;               // 16-col tile of HS
    int hm = blockIdx.z;               // head*3 + which
    int head = hm / 3, which = hm % 3;

    const bf16* w  = wt + (((long)which * HH + head) * HSZ + nt * 16) * CC;
    const bf16* A0 = h + (long)(mt * 16) * CC;

    v8f acc = {};
    for (int kk = 0; kk < CC; kk += 32) {
        v16bf a = load_a16x32(A0 + kk, CC);
        v16bf b = load_b32x16_t(w + kk, CC);
        acc = wmma_acc(a, b, acc);
    }

    int lane = threadIdx.x & 31;
    int dcol = nt * 16 + (lane & 15);
    const float* bias = (which == 0) ? bq : (which == 1) ? bk : bv;
    float bia = bias[head * HSZ + dcol];

    int bidx = (mt * 16) / TT, t0 = (mt * 16) % TT;
    if (which < 2) {
        bf16* out = ((which == 0) ? q : k) +
                    (((long)bidx * HH + head) * TT + t0) * HSZ;
#pragma unroll
        for (int i = 0; i < 8; ++i) {
            int M = i + ((lane >> 4) << 3);
            out[(long)M * HSZ + dcol] = (bf16)(acc[i] + bia);
        }
    } else {
        bf16* out = vT + (((long)bidx * HH + head) * HSZ + dcol) * TT + t0;
#pragma unroll
        for (int i = 0; i < 8; ++i) {
            int M = i + ((lane >> 4) << 3);
            out[M] = (bf16)(acc[i] + bia);
        }
    }
}

// ---------------------------------------------------------------------------
// 5) Causal flash attention. One wave per (b, head, 16-row query tile).
//    scores = Q K^T / sqrt(C); online softmax; O = P V; y concat-heads bf16.
// ---------------------------------------------------------------------------
__global__ void k_attn(const bf16* __restrict__ q, const bf16* __restrict__ k,
                       const bf16* __restrict__ vT, bf16* __restrict__ y) {
    int mt   = blockIdx.x;   // query tile (T/16)
    int head = blockIdx.y;
    int b    = blockIdx.z;
    int lane = threadIdx.x & 31;

    const bf16* qb = q  + ((long)b * HH + head) * TT * HSZ;
    const bf16* kb = k  + ((long)b * HH + head) * TT * HSZ;
    const bf16* vb = vT + ((long)b * HH + head) * HSZ * TT;

    __shared__ bf16 pl[16][32];

    int tbase = mt * 16;
    v16bf aq0 = load_a16x32(qb + (long)tbase * HSZ + 0,  HSZ);
    v16bf aq1 = load_a16x32(qb + (long)tbase * HSZ + 32, HSZ);

    v8f o0 = {}, o1 = {}, o2 = {}, o3 = {};
    float mrun[8], lrun[8];
    int mrow[8];
#pragma unroll
    for (int i = 0; i < 8; ++i) {
        mrow[i] = i + ((lane >> 4) << 3);
        mrun[i] = -1e30f;
        lrun[i] = 0.f;
    }
    int ncol = lane & 15;
    const float scale = 0.03125f;            // C^-0.5 = 1/32 (per reference)
    int scmax = (tbase + 15) >> 5;           // causal bound on 32-wide chunks

    for (int sc = 0; sc <= scmax; ++sc) {
        int s0 = sc << 5;
        // ---- scores: two 16x16 tiles over K=HS=64 (two wmma each) ----
        v8f sA = {}, sB = {};
        sA = wmma_acc(aq0, load_b32x16_t(kb + (long)s0 * HSZ + 0,  HSZ), sA);
        sA = wmma_acc(aq1, load_b32x16_t(kb + (long)s0 * HSZ + 32, HSZ), sA);
        sB = wmma_acc(aq0, load_b32x16_t(kb + (long)(s0 + 16) * HSZ + 0,  HSZ), sB);
        sB = wmma_acc(aq1, load_b32x16_t(kb + (long)(s0 + 16) * HSZ + 32, HSZ), sB);

        // ---- mask + online softmax (row stats via half-wave shuffles) ----
        float pA[8], pB[8], alpha[8];
#pragma unroll
        for (int i = 0; i < 8; ++i) {
            int tg = tbase + mrow[i];
            float a_ = sA[i] * scale;
            float b_ = sB[i] * scale;
            if (s0 + ncol > tg)      a_ = -1e30f;
            if (s0 + 16 + ncol > tg) b_ = -1e30f;
            float rm = fmaxf(a_, b_);
#pragma unroll
            for (int m = 1; m < 16; m <<= 1) rm = fmaxf(rm, __shfl_xor(rm, m, 16));
            float mnew = fmaxf(mrun[i], rm);
            alpha[i] = __expf(mrun[i] - mnew);
            pA[i] = __expf(a_ - mnew);
            pB[i] = __expf(b_ - mnew);
            float rs = pA[i] + pB[i];
#pragma unroll
            for (int m = 1; m < 16; m <<= 1) rs += __shfl_xor(rs, m, 16);
            lrun[i] = lrun[i] * alpha[i] + rs;
            mrun[i] = mnew;
        }
#pragma unroll
        for (int i = 0; i < 8; ++i) {
            o0[i] *= alpha[i]; o1[i] *= alpha[i];
            o2[i] *= alpha[i]; o3[i] *= alpha[i];
        }

        // ---- transpose P (C-layout -> A-layout) through LDS ----
        __syncthreads();
#pragma unroll
        for (int i = 0; i < 8; ++i) {
            pl[mrow[i]][ncol]      = (bf16)pA[i];
            pl[mrow[i]][16 + ncol] = (bf16)pB[i];
        }
        __syncthreads();
        v16bf ap = load_a16x32(&pl[0][0], 32);

        // ---- O += P * V  (V pre-transposed -> contiguous B fragments) ----
        o0 = wmma_acc(ap, load_b32x16_t(vb + (long)0  * TT + s0, TT), o0);
        o1 = wmma_acc(ap, load_b32x16_t(vb + (long)16 * TT + s0, TT), o1);
        o2 = wmma_acc(ap, load_b32x16_t(vb + (long)32 * TT + s0, TT), o2);
        o3 = wmma_acc(ap, load_b32x16_t(vb + (long)48 * TT + s0, TT), o3);
    }

    // ---- finalize and write concat-head y[b,t,head*64+d] ----
    bf16* yb = y + (long)b * TT * CC + head * HSZ;
#pragma unroll
    for (int i = 0; i < 8; ++i) {
        float inv = 1.f / lrun[i];
        bf16* row = yb + (long)(tbase + mrow[i]) * CC;
        row[0  + ncol] = (bf16)(o0[i] * inv);
        row[16 + ncol] = (bf16)(o1[i] * inv);
        row[32 + ncol] = (bf16)(o2[i] * inv);
        row[48 + ncol] = (bf16)(o3[i] * inv);
    }
}

// ---------------------------------------------------------------------------
// 6) LM head GEMM: logits[4096,32000] = y[4096,1024] * Wt^T + b  (fp32 out)
//    One wave per 16x64 tile; weight pre-transposed [V][C] bf16.
// ---------------------------------------------------------------------------
__global__ void k_head(const bf16* __restrict__ y, const bf16* __restrict__ wt,
                       const float* __restrict__ bias, float* __restrict__ out) {
    int mt = blockIdx.x;               // 16-row tile (BT/16)
    int nt = blockIdx.y;               // 64-col tile (V/64)
    int lane = threadIdx.x & 31;

    const bf16* A0 = y  + (long)(mt * 16) * CC;
    const bf16* W0 = wt + (long)(nt * 64) * CC;

    v8f a0 = {}, a1 = {}, a2 = {}, a3 = {};
    for (int kk = 0; kk < CC; kk += 32) {
        v16bf a = load_a16x32(A0 + kk, CC);
        a0 = wmma_acc(a, load_b32x16_t(W0 + (long)0  * CC + kk, CC), a0);
        a1 = wmma_acc(a, load_b32x16_t(W0 + (long)16 * CC + kk, CC), a1);
        a2 = wmma_acc(a, load_b32x16_t(W0 + (long)32 * CC + kk, CC), a2);
        a3 = wmma_acc(a, load_b32x16_t(W0 + (long)48 * CC + kk, CC), a3);
    }

    int ncol = lane & 15;
    float b0 = bias[nt * 64 + 0  + ncol];
    float b1 = bias[nt * 64 + 16 + ncol];
    float b2 = bias[nt * 64 + 32 + ncol];
    float b3 = bias[nt * 64 + 48 + ncol];
#pragma unroll
    for (int i = 0; i < 8; ++i) {
        int M = i + ((lane >> 4) << 3);
        float* row = out + (long)(mt * 16 + M) * VV + nt * 64;
        row[0  + ncol] = a0[i] + b0;
        row[16 + ncol] = a1[i] + b1;
        row[32 + ncol] = a2[i] + b2;
        row[48 + ncol] = a3[i] + b3;
    }
}

// ---------------------------------------------------------------------------
extern "C" void kernel_launch(void* const* d_in, const int* in_sizes, int n_in,
                              void* d_out, int out_size, void* d_ws, size_t ws_size,
                              hipStream_t stream) {
    const int*   x      = (const int*)  d_in[0];
    const float* tok    = (const float*)d_in[1];
    const float* pos    = (const float*)d_in[2];
    const float* wq     = (const float*)d_in[3];
    const float* bq     = (const float*)d_in[4];
    const float* wk     = (const float*)d_in[5];
    const float* bk     = (const float*)d_in[6];
    const float* wv     = (const float*)d_in[7];
    const float* bv     = (const float*)d_in[8];
    const float* w_head = (const float*)d_in[9];
    const float* b_head = (const float*)d_in[10];
    (void)in_sizes; (void)n_in; (void)out_size; (void)ws_size;

    char* ws = (char*)d_ws;
    size_t off = 0;
    auto carve = [&](size_t bytes) -> char* {
        char* p = ws + off;
        off = (off + bytes + 255) & ~(size_t)255;
        return p;
    };
    bf16* h      = (bf16*)carve((size_t)BT * CC * 2);
    bf16* wqkv_t = (bf16*)carve((size_t)3 * HH * HSZ * CC * 2);
    bf16* qb     = (bf16*)carve((size_t)BB * HH * TT * HSZ * 2);
    bf16* kbuf   = (bf16*)carve((size_t)BB * HH * TT * HSZ * 2);
    bf16* vT     = (bf16*)carve((size_t)BB * HH * HSZ * TT * 2);
    bf16* yb     = (bf16*)carve((size_t)BT * CC * 2);
    bf16* wh_t   = (bf16*)carve((size_t)VV * CC * 2);

    bf16* wq_t = wqkv_t;
    bf16* wk_t = wqkv_t + (size_t)HH * HSZ * CC;
    bf16* wv_t = wqkv_t + (size_t)2 * HH * HSZ * CC;

    // 1) embed + cast
    k_embed<<<BT, 256, 0, stream>>>(x, tok, pos, h);

    // 2) weight cast/transposes
    int nW = HH * HSZ * CC;
    k_cast_wqkv<<<(nW + 255) / 256, 256, 0, stream>>>(wq, wq_t);
    k_cast_wqkv<<<(nW + 255) / 256, 256, 0, stream>>>(wk, wk_t);
    k_cast_wqkv<<<(nW + 255) / 256, 256, 0, stream>>>(wv, wv_t);
    k_cast_whead<<<dim3(VV / 64, CC / 64), 256, 0, stream>>>(w_head, wh_t);

    // 3) QKV projection (WMMA)
    k_qkv<<<dim3(BT / 16, HSZ / 16, HH * 3), 32, 0, stream>>>(
        h, wqkv_t, bq, bk, bv, qb, kbuf, vT);

    // 4) causal flash attention (WMMA)
    k_attn<<<dim3(TT / 16, HH, BB), 32, 0, stream>>>(qb, kbuf, vT, yb);

    // 5) LM head GEMM (WMMA, memory-bound: 524MB logits + 64MB bf16 weight)
    k_head<<<dim3(BT / 16, VV / 64), 32, 0, stream>>>(yb, wh_t, b_head,
                                                      (float*)d_out);
}
